// AnswerLayer_15032385536360
// MI455X (gfx1250) — matrix operand order repaired and checked by reading, not compiled, not run on hardware
//
#include <hip/hip_runtime.h>
#include <hip/hip_bf16.h>

#define B_     32
#define L_     512
#define D_     1024
#define TURNS_ 5

typedef __attribute__((ext_vector_type(16))) _Float16 v16h;
typedef __attribute__((ext_vector_type(8)))  _Float16 v8h;
typedef __attribute__((ext_vector_type(8)))  float    v8f;

// ---------------------------------------------------------------------------
// fp32 -> f16 conversion (weights once per launch; activations once per use)
// ---------------------------------------------------------------------------
__global__ void al_cvt_f16(const float* __restrict__ src,
                           _Float16* __restrict__ dst, int n) {
  int i = blockIdx.x * blockDim.x + threadIdx.x;
  if (i < n) dst[i] = (_Float16)src[i];
}

// ---------------------------------------------------------------------------
// WMMA GEMM:  C[32, N] = A[32, K](f16) @ W[N, K]^T (f16)  (+bias, +addend)
// One wave computes a 16x64 output strip: 1 A-fragment feeds 4 WMMAs.
// Grid: (N/64, 32/16), block = 32 (one wave).
// Fragment packing used here (contiguous K, same for A and B; 16-bit A 16x32 / B 32x16):
//   A: lanes 0-15 row M=l, elems 0..15 = K+0..15;
//      lanes 16-31 row M=l-16, K+16..31.
//   B: lane l holds column n = l&15; lanes<16 hold K..K+15, lanes>=16
//      K+16..K+31 (contiguous in W's K-major rows -> one 32B load).
// ---------------------------------------------------------------------------
__global__ __launch_bounds__(32) void al_gemm_a32_wmma(
    const _Float16* __restrict__ A,      // [32, K] f16 row-major
    const _Float16* __restrict__ W,      // [N, K]  f16 row-major (use W^T)
    const float*    __restrict__ bias,   // [N] or nullptr
    const float*    __restrict__ addend, // [32, N] or nullptr
    float*          __restrict__ C,      // [32, N]
    int N, int K)
{
  const int lane = threadIdx.x;        // 0..31
  const int hgrp = lane >> 4;          // 0 | 1 (lane half)
  const int l16  = lane & 15;
  const int tn0  = blockIdx.x * 4;     // first of four 16-wide N tiles
  const int tm   = blockIdx.y;         // M tile (0..1)

  const _Float16* Arow = A + (size_t)(tm * 16 + l16) * K;
  const _Float16* Wc0  = W + (size_t)(tn0 * 16 + l16) * K; // col for tile 0
  const size_t    wstr = (size_t)16 * K;                   // tile-to-tile

  v8f acc[4] = {v8f{}, v8f{}, v8f{}, v8f{}};
  for (int k0 = 0; k0 < K; k0 += 32) {
    // A fragment: one 32B load, contiguous K (matches B)
    v16h a = *(const v16h*)(Arow + k0 + hgrp * 16);
    const int koff = k0 + hgrp * 16;
    v16h b0 = *(const v16h*)(Wc0 + 0 * wstr + koff);
    v16h b1 = *(const v16h*)(Wc0 + 1 * wstr + koff);
    v16h b2 = *(const v16h*)(Wc0 + 2 * wstr + koff);
    v16h b3 = *(const v16h*)(Wc0 + 3 * wstr + koff);

    acc[0] = __builtin_amdgcn_wmma_f32_16x16x32_f16(false, a, false, b0,
                                                    (short)0, acc[0], false, false);
    acc[1] = __builtin_amdgcn_wmma_f32_16x16x32_f16(false, a, false, b1,
                                                    (short)0, acc[1], false, false);
    acc[2] = __builtin_amdgcn_wmma_f32_16x16x32_f16(false, a, false, b2,
                                                    (short)0, acc[2], false, false);
    acc[3] = __builtin_amdgcn_wmma_f32_16x16x32_f16(false, a, false, b3,
                                                    (short)0, acc[3], false, false);
  }

  // C/D layout: VGPR v -> row (tm*16 + hgrp*8 + v), col = tile*16 + (lane&15)
  const int crow0 = tm * 16 + hgrp * 8;
#pragma unroll
  for (int t = 0; t < 4; ++t) {
    const int   ncol = (tn0 + t) * 16 + l16;
    const float badd = bias ? bias[ncol] : 0.0f;
#pragma unroll
    for (int v = 0; v < 8; ++v) {
      size_t ci  = (size_t)(crow0 + v) * N + ncol;
      float  val = acc[t][v] + badd;
      if (addend) val += addend[ci];
      C[ci] = val;
    }
  }
}

// ---------------------------------------------------------------------------
// Bilinear scores: out[b,l] = dot(M[b,l,:], u[b,:]) + bias_scalar
// One wave per (b,l) row; float4 loads; wave-shuffle reduction.
// ---------------------------------------------------------------------------
__global__ __launch_bounds__(256) void al_bilin_scores(
    const float* __restrict__ M, const float* __restrict__ u,
    const float* __restrict__ bias_scalar, float* __restrict__ out)
{
  const int gwave = (blockIdx.x * blockDim.x + threadIdx.x) >> 5; // row id
  const int lane  = threadIdx.x & 31;
  const int b = gwave / L_;
  const int l = gwave % L_;

  const float4* Mr = (const float4*)(M + ((size_t)b * L_ + l) * D_);
  const float4* ur = (const float4*)(u + (size_t)b * D_);

  // gfx1250 speculative prefetch of the next memory row (global_prefetch_b8)
  __builtin_prefetch(((const char*)Mr) + (size_t)D_ * sizeof(float), 0, 1);

  float acc = 0.0f;
#pragma unroll
  for (int j = 0; j < D_ / 4 / 32; ++j) {   // 8 iterations
    float4 m = Mr[lane + j * 32];
    float4 q = ur[lane + j * 32];
    acc += m.x * q.x + m.y * q.y + m.z * q.z + m.w * q.w;
  }
#pragma unroll
  for (int off = 16; off > 0; off >>= 1) acc += __shfl_xor(acc, off, 32);
  if (lane == 0) out[(size_t)b * L_ + l] = acc + bias_scalar[0];
}

// ---------------------------------------------------------------------------
// Softmax over L=512 per batch row (one block per b, 512 threads)
// ---------------------------------------------------------------------------
__global__ __launch_bounds__(512) void al_softmax(
    const float* __restrict__ scores, float* __restrict__ att)
{
  __shared__ float red[L_];
  const int b = blockIdx.x;
  const int t = threadIdx.x;
  const float v = scores[(size_t)b * L_ + t];

  red[t] = v;
  __syncthreads();
  for (int s = L_ / 2; s > 0; s >>= 1) {
    if (t < s) red[t] = fmaxf(red[t], red[t + s]);
    __syncthreads();
  }
  const float m = red[0];
  __syncthreads();

  const float e = __expf(v - m);
  red[t] = e;
  __syncthreads();
  for (int s = L_ / 2; s > 0; s >>= 1) {
    if (t < s) red[t] += red[t + s];
    __syncthreads();
  }
  att[(size_t)b * L_ + t] = e / red[0];
}

// ---------------------------------------------------------------------------
// Attention readout: x[b,d] = sum_l att[b,l] * M[b,l,d]
// Grid (B, D/256), 256 threads; att row staged in LDS, coalesced along d.
// ---------------------------------------------------------------------------
__global__ __launch_bounds__(256) void al_readout(
    const float* __restrict__ M, const float* __restrict__ att,
    float* __restrict__ x)
{
  __shared__ float a_sh[L_];
  const int b = blockIdx.x;
  const int d = blockIdx.y * blockDim.x + threadIdx.x;
  // cooperative stage of the attention row (512 floats, 2 per thread)
  a_sh[threadIdx.x]       = att[(size_t)b * L_ + threadIdx.x];
  a_sh[threadIdx.x + 256] = att[(size_t)b * L_ + threadIdx.x + 256];
  __syncthreads();

  const float* Mb = M + (size_t)b * L_ * D_;
  float acc = 0.0f;
  for (int l = 0; l < L_; ++l) acc = fmaf(a_sh[l], Mb[(size_t)l * D_ + d], acc);
  x[(size_t)b * D_ + d] = acc;
}

// ---------------------------------------------------------------------------
// GRU gates: r=sig(gi_r+gh_r); z=sig(gi_z+gh_z); n=tanh(gi_n + r*gh_n);
//            s' = (1-z)*n + z*s
// ---------------------------------------------------------------------------
__global__ __launch_bounds__(256) void al_gru_gate(
    const float* __restrict__ gi, const float* __restrict__ gh,
    const float* __restrict__ s, float* __restrict__ s_out)
{
  const int i = blockIdx.x * blockDim.x + threadIdx.x; // over B*D
  const int b = i / D_;
  const int d = i % D_;
  const float* gib = gi + (size_t)b * 3 * D_;
  const float* ghb = gh + (size_t)b * 3 * D_;
  const float r = 1.0f / (1.0f + __expf(-(gib[d] + ghb[d])));
  const float z = 1.0f / (1.0f + __expf(-(gib[D_ + d] + ghb[D_ + d])));
  const float n = tanhf(gib[2 * D_ + d] + r * ghb[2 * D_ + d]);
  s_out[i] = (1.0f - z) * n + z * s[i];
}

// ---------------------------------------------------------------------------
// Host-side orchestration
// ---------------------------------------------------------------------------
extern "C" void kernel_launch(void* const* d_in, const int* in_sizes, int n_in,
                              void* d_out, int out_size, void* d_ws,
                              size_t ws_size, hipStream_t stream) {
  (void)in_sizes; (void)n_in; (void)out_size; (void)ws_size;

  const float* M      = (const float*)d_in[0];
  const float* s0     = (const float*)d_in[1];
  const float* Wstart = (const float*)d_in[2];
  const float* bstart = (const float*)d_in[3];  // scalar
  const float* Wend   = (const float*)d_in[4];
  const float* bend   = (const float*)d_in[5];  // scalar
  const float* Wproj  = (const float*)d_in[6];
  const float* bproj  = (const float*)d_in[7];  // [D]
  const float* Wih    = (const float*)d_in[8];  // [3D, D]
  const float* Whh    = (const float*)d_in[9];  // [3D, D]
  const float* bih    = (const float*)d_in[10]; // [3D]
  const float* bhh    = (const float*)d_in[11]; // [3D]

  float* out_starts = (float*)d_out;                           // [T, B, L]
  float* out_ends   = out_starts + (size_t)TURNS_ * B_ * L_;   // [T, B, L]

  // ---- carve workspace (~21 MB) ----
  char*  ws  = (char*)d_ws;
  size_t off = 0;
  auto carve = [&](size_t bytes) -> void* {
    void* p = ws + off;
    off += (bytes + 255) & ~(size_t)255;
    return p;
  };
  _Float16* Wsh  = (_Float16*)carve((size_t)D_ * D_ * sizeof(_Float16));
  _Float16* Weh  = (_Float16*)carve((size_t)D_ * D_ * sizeof(_Float16));
  _Float16* Wph  = (_Float16*)carve((size_t)D_ * D_ * sizeof(_Float16));
  _Float16* Wihh = (_Float16*)carve((size_t)3 * D_ * D_ * sizeof(_Float16));
  _Float16* Whhh = (_Float16*)carve((size_t)3 * D_ * D_ * sizeof(_Float16));
  _Float16* sh   = (_Float16*)carve((size_t)B_ * D_ * sizeof(_Float16));
  _Float16* xh   = (_Float16*)carve((size_t)B_ * D_ * sizeof(_Float16));
  _Float16* qh   = (_Float16*)carve((size_t)B_ * D_ * sizeof(_Float16));
  float* sbuf0 = (float*)carve((size_t)B_ * D_ * sizeof(float));
  float* sbuf1 = (float*)carve((size_t)B_ * D_ * sizeof(float));
  float* u     = (float*)carve((size_t)B_ * D_ * sizeof(float));
  float* att   = (float*)carve((size_t)B_ * L_ * sizeof(float));
  float* x     = (float*)carve((size_t)B_ * D_ * sizeof(float));
  float* q     = (float*)carve((size_t)B_ * D_ * sizeof(float));
  float* wendv = (float*)carve((size_t)B_ * D_ * sizeof(float));
  float* gi    = (float*)carve((size_t)B_ * 3 * D_ * sizeof(float));
  float* gh    = (float*)carve((size_t)B_ * 3 * D_ * sizeof(float));
  float* sb[2] = {sbuf0, sbuf1};

  // ---- weight conversion fp32 -> f16 (deterministic, every launch) ----
  const int nDD  = D_ * D_;
  const int n3DD = 3 * D_ * D_;
  const int nBD  = B_ * D_;
  al_cvt_f16<<<(nDD  + 255) / 256, 256, 0, stream>>>(Wstart, Wsh,  nDD);
  al_cvt_f16<<<(nDD  + 255) / 256, 256, 0, stream>>>(Wend,   Weh,  nDD);
  al_cvt_f16<<<(nDD  + 255) / 256, 256, 0, stream>>>(Wproj,  Wph,  nDD);
  al_cvt_f16<<<(n3DD + 255) / 256, 256, 0, stream>>>(Wih,    Wihh, n3DD);
  al_cvt_f16<<<(n3DD + 255) / 256, 256, 0, stream>>>(Whh,    Whhh, n3DD);

  hipMemcpyAsync(sb[0], s0, (size_t)B_ * D_ * sizeof(float),
                 hipMemcpyDeviceToDevice, stream);

  int cur = 0;
  for (int t = 0; t < TURNS_; ++t) {
    float* st_out = out_starts + (size_t)t * B_ * L_;
    float* en_out = out_ends   + (size_t)t * B_ * L_;

    // s -> f16 (used by u-GEMM and gh-GEMM)
    al_cvt_f16<<<(nBD + 255) / 256, 256, 0, stream>>>(sb[cur], sh, nBD);
    // u = s @ W_start^T
    al_gemm_a32_wmma<<<dim3(D_ / 64, B_ / 16), 32, 0, stream>>>(
        sh, Wsh, nullptr, nullptr, u, D_, D_);
    // start[b,l] = M[b,l,:].u[b,:] + b_start
    al_bilin_scores<<<(B_ * L_) / 8, 256, 0, stream>>>(M, u, bstart, st_out);
    // att = softmax_L(start)
    al_softmax<<<B_, L_, 0, stream>>>(st_out, att);
    // x = att^T M
    al_readout<<<dim3(B_, D_ / 256), 256, 0, stream>>>(M, att, x);
    // x -> f16 (used by proj-GEMM and gi-GEMM)
    al_cvt_f16<<<(nBD + 255) / 256, 256, 0, stream>>>(x, xh, nBD);
    // q = s + x @ W_proj^T + b_proj
    al_gemm_a32_wmma<<<dim3(D_ / 64, B_ / 16), 32, 0, stream>>>(
        xh, Wph, bproj, sb[cur], q, D_, D_);
    // q -> f16
    al_cvt_f16<<<(nBD + 255) / 256, 256, 0, stream>>>(q, qh, nBD);
    // w = q @ W_end^T
    al_gemm_a32_wmma<<<dim3(D_ / 64, B_ / 16), 32, 0, stream>>>(
        qh, Weh, nullptr, nullptr, wendv, D_, D_);
    // end[b,l] = M[b,l,:].w[b,:] + b_end
    al_bilin_scores<<<(B_ * L_) / 8, 256, 0, stream>>>(M, wendv, bend, en_out);
    // GRU: gi = x @ W_ih^T + b_ih ; gh = s @ W_hh^T + b_hh
    al_gemm_a32_wmma<<<dim3(3 * D_ / 64, B_ / 16), 32, 0, stream>>>(
        xh, Wihh, bih, nullptr, gi, 3 * D_, D_);
    al_gemm_a32_wmma<<<dim3(3 * D_ / 64, B_ / 16), 32, 0, stream>>>(
        sh, Whhh, bhh, nullptr, gh, 3 * D_, D_);
    // gates -> s'
    al_gru_gate<<<(B_ * D_) / 256, 256, 0, stream>>>(gi, gh, sb[cur],
                                                     sb[1 - cur]);
    cur ^= 1;
  }
}
